// FusedMoELayer_48576080118266
// MI455X (gfx1250) — matrix operand order repaired and checked by reading, not compiled, not run on hardware
//
#include <hip/hip_runtime.h>
#include <hip/hip_bf16.h>
#include <math.h>

#define D_MODEL 1024
#define D_FF    4096
#define NE      8
#define NTOK    4096                      // 2 * 2048
#define BM      32                        // pair rows per block
#define BN      128                       // N columns per block (8 waves x 16)
#define KC      64                        // K elements staged per iteration
#define LDSTR   72                        // padded LDS row stride (bf16 units)
#define NPAIR_CAP (NTOK * 2 + NE * BM)

typedef __attribute__((ext_vector_type(16))) __bf16 v16bf;
typedef __attribute__((ext_vector_type(8)))  __bf16 v8bf;
typedef __attribute__((ext_vector_type(8)))  float  v8f;

// meta ints: [0..7] counts, [8..15] cursors, [16..24] offsets

// ---------------------------------------------------------------------------
__global__ __launch_bounds__(256) void k_router(const float* __restrict__ x,
                                                const float* __restrict__ rw,
                                                int* __restrict__ topk_id,
                                                float* __restrict__ topk_w,
                                                int* __restrict__ meta) {
    __shared__ float s_rw[NE * D_MODEL];
    for (int i = threadIdx.x; i < NE * D_MODEL; i += 256) s_rw[i] = rw[i];
    __syncthreads();

    const int wv = threadIdx.x >> 5;
    const int lane = threadIdx.x & 31;
    const int n = blockIdx.x * 8 + wv;

    float xr[32];
    const float* xp = x + (size_t)n * D_MODEL + lane;
#pragma unroll
    for (int i = 0; i < 32; ++i) xr[i] = xp[i * 32];

    float lg[NE];
#pragma unroll
    for (int e = 0; e < NE; ++e) {
        float a = 0.f;
#pragma unroll
        for (int i = 0; i < 32; ++i) a += xr[i] * s_rw[e * D_MODEL + lane + i * 32];
#pragma unroll
        for (int o = 16; o > 0; o >>= 1) a += __shfl_xor(a, o, 32);
        lg[e] = a;
    }

    int b0 = 0, b1 = -1;
    float v0 = lg[0], v1 = -3.4e38f;
#pragma unroll
    for (int e = 1; e < NE; ++e) {
        if (lg[e] > v0) { v1 = v0; b1 = b0; v0 = lg[e]; b0 = e; }
        else if (lg[e] > v1) { v1 = lg[e]; b1 = e; }
    }
    float p0 = 1.f / (1.f + __expf(v1 - v0));

    if (lane == 0) {
        topk_id[2 * n] = b0;  topk_w[2 * n] = p0;
        topk_id[2 * n + 1] = b1;  topk_w[2 * n + 1] = 1.f - p0;
        atomicAdd(&meta[b0], 1);
        atomicAdd(&meta[b1], 1);
    }
}

// Exclusive scan (pad each expert to BM rows) + init pad pair records.
__global__ void k_scan(int* __restrict__ meta,
                       int* __restrict__ pair_tok,
                       float* __restrict__ pair_w) {
    if (threadIdx.x == 0) {
        const int* counts = meta;
        int* offsets = meta + 16;
        int off = 0;
        for (int e = 0; e < NE; ++e) {
            offsets[e] = off;
            int c = counts[e];
            int padded = (c + BM - 1) & ~(BM - 1);
            for (int i = c; i < padded; ++i) { pair_tok[off + i] = 0; pair_w[off + i] = 0.f; }
            off += padded;
        }
        offsets[NE] = off;
    }
}

__global__ __launch_bounds__(256) void k_scatter(const int* __restrict__ topk_id,
                                                 const float* __restrict__ topk_w,
                                                 int* __restrict__ meta,
                                                 int* __restrict__ pair_tok,
                                                 float* __restrict__ pair_w) {
    int n = blockIdx.x * 256 + threadIdx.x;
    const int* offsets = meta + 16;
    int* cursors = meta + 8;
#pragma unroll
    for (int k = 0; k < 2; ++k) {
        int e = topk_id[2 * n + k];
        int slot = offsets[e] + atomicAdd(&cursors[e], 1);
        pair_tok[slot] = n;
        pair_w[slot] = topk_w[2 * n + k];
    }
}

// ---------------------------------------------------------------------------
// GEMM1: H = gelu(x . w1^T), grouped by expert. 256 thr (8 waves),
// 32 pairs x 128 F per block, KC=64 double-buffered bf16 LDS staging,
// 4 v_wmma_f32_16x16x32_bf16 per stage per wave (2 K-sub x 2 M-tiles).
// ---------------------------------------------------------------------------
__global__ __launch_bounds__(256) void k_gemm1(const float* __restrict__ x,
                                               const float* __restrict__ w1,
                                               const int* __restrict__ meta,
                                               const int* __restrict__ pair_tok,
                                               __bf16* __restrict__ H) {
    const int* offsets = meta + 16;
    const int e = blockIdx.y;
    int rowsStart = offsets[e];
    const int padded = offsets[e + 1] - rowsStart;
    if ((int)blockIdx.x * BM >= padded) return;
    rowsStart += blockIdx.x * BM;
    const int fBase = blockIdx.z * BN;
    const float* w1e = w1 + (size_t)e * D_FF * D_MODEL;

    __shared__ int s_tok[BM];
    alignas(16) __shared__ __bf16 s_x[2][BM * LDSTR];
    alignas(16) __shared__ __bf16 s_w[2][BN * LDSTR];
    if (threadIdx.x < BM) s_tok[threadIdx.x] = pair_tok[rowsStart + threadIdx.x];
    __syncthreads();

    const int t = threadIdx.x;
    const int xr = t >> 3, xc = (t & 7) * 8;     // X stage: 8 fp32 / thread
    const int wr = t >> 1, wc = (t & 1) * 32;    // W stage: 32 fp32 / thread
    const float* xp = x + (size_t)s_tok[xr] * D_MODEL + xc;     // hoisted token ptr
    const float* wp = w1e + (size_t)(fBase + wr) * D_MODEL + wc;

    const int lane = t & 31, wv = t >> 5;
    const int m = lane & 15, hi = lane >> 4;

    v8f acc[2];
    acc[0] = (v8f){0.f,0.f,0.f,0.f,0.f,0.f,0.f,0.f};
    acc[1] = acc[0];

    float4 xq[2], wq[8];
    auto loadStage = [&](int kb) {
#pragma unroll
        for (int j = 0; j < 2; ++j) xq[j] = ((const float4*)(xp + kb))[j];
#pragma unroll
        for (int j = 0; j < 8; ++j) wq[j] = ((const float4*)(wp + kb))[j];
    };
    auto storeStage = [&](int b) {
        v8bf vx;
#pragma unroll
        for (int j = 0; j < 2; ++j) {
            vx[j*4+0] = (__bf16)xq[j].x; vx[j*4+1] = (__bf16)xq[j].y;
            vx[j*4+2] = (__bf16)xq[j].z; vx[j*4+3] = (__bf16)xq[j].w;
        }
        *(v8bf*)&s_x[b][xr * LDSTR + xc] = vx;
#pragma unroll
        for (int c = 0; c < 4; ++c) {
            v8bf vw;
#pragma unroll
            for (int j = 0; j < 2; ++j) {
                float4 q = wq[c*2+j];
                vw[j*4+0] = (__bf16)q.x; vw[j*4+1] = (__bf16)q.y;
                vw[j*4+2] = (__bf16)q.z; vw[j*4+3] = (__bf16)q.w;
            }
            *(v8bf*)&s_w[b][wr * LDSTR + wc + c * 8] = vw;
        }
    };

    loadStage(0);
    storeStage(0);
    __syncthreads();

    const int NS = D_MODEL / KC;
    for (int s = 0; s < NS; ++s) {
        const int b = s & 1;
        if (s + 1 < NS) loadStage((s + 1) * KC);
#pragma unroll
        for (int ks = 0; ks < 2; ++ks) {
            const int kc = ks * 32;
            v8bf bb0 = *(const v8bf*)&s_w[b][(wv*16 + m) * LDSTR + kc + hi*8];
            v8bf bb1 = *(const v8bf*)&s_w[b][(wv*16 + m) * LDSTR + kc + 16 + hi*8];
            v16bf B = __builtin_shufflevector(bb0, bb1, 0,1,2,3,4,5,6,7,8,9,10,11,12,13,14,15);
#pragma unroll
            for (int mt = 0; mt < 2; ++mt) {
                v8bf aa0 = *(const v8bf*)&s_x[b][(mt*16 + m) * LDSTR + kc + hi*8];
                v8bf aa1 = *(const v8bf*)&s_x[b][(mt*16 + m) * LDSTR + kc + 16 + hi*8];
                v16bf A = __builtin_shufflevector(aa0, aa1, 0,1,2,3,4,5,6,7,8,9,10,11,12,13,14,15);
                acc[mt] = __builtin_amdgcn_wmma_f32_16x16x32_bf16(false, A, false, B,
                                                                 (short)0, acc[mt], false, false);
            }
        }
        if (s + 1 < NS) storeStage((s + 1) & 1);
        __syncthreads();
    }

#pragma unroll
    for (int mt = 0; mt < 2; ++mt)
#pragma unroll
        for (int r = 0; r < 8; ++r) {
            const int M = mt * 16 + r + hi * 8;
            float v = acc[mt][r];
            float g = 0.5f * v * (1.0f + erff(v * 0.70710678118654752f));
            H[(size_t)(rowsStart + M) * D_FF + fBase + wv * 16 + m] = (__bf16)g;
        }
}

// ---------------------------------------------------------------------------
// GEMM2: out += w(p) * (H . w2^T). A-tile (bf16 H) is DMA'd straight into LDS
// with GLOBAL_LOAD_ASYNC_TO_LDS_B128 (ASYNCcnt); W-tile converted fp32->bf16.
// ---------------------------------------------------------------------------
__global__ __launch_bounds__(256) void k_gemm2(const __bf16* __restrict__ H,
                                               const float* __restrict__ w2,
                                               const int* __restrict__ meta,
                                               const int* __restrict__ pair_tok,
                                               const float* __restrict__ pair_w,
                                               float* __restrict__ out) {
    const int* offsets = meta + 16;
    const int e = blockIdx.y;
    int rowsStart = offsets[e];
    const int padded = offsets[e + 1] - rowsStart;
    if ((int)blockIdx.x * BM >= padded) return;
    rowsStart += blockIdx.x * BM;
    const int dBase = blockIdx.z * BN;
    const float* w2e = w2 + (size_t)e * D_MODEL * D_FF;

    __shared__ int s_tok[BM];
    __shared__ float s_pw[BM];
    alignas(16) __shared__ __bf16 s_x[2][BM * LDSTR];
    alignas(16) __shared__ __bf16 s_w[2][BN * LDSTR];
    if (threadIdx.x < BM) {
        s_tok[threadIdx.x] = pair_tok[rowsStart + threadIdx.x];
        s_pw[threadIdx.x]  = pair_w[rowsStart + threadIdx.x];
    }
    __syncthreads();

    const int t = threadIdx.x;
    const int xr = t >> 3, xc = (t & 7) * 8;     // A stage: one B128 DMA / thread
    const int wr = t >> 1, wc = (t & 1) * 32;
    const __bf16* hp = H + (size_t)(rowsStart + xr) * D_FF + xc;
    const float* wp = w2e + (size_t)(dBase + wr) * D_FF + wc;
    const unsigned ldsX[2] = { (unsigned)(uintptr_t)&s_x[0][xr * LDSTR + xc],
                               (unsigned)(uintptr_t)&s_x[1][xr * LDSTR + xc] };

    const int lane = t & 31, wv = t >> 5;
    const int m = lane & 15, hi = lane >> 4;

    v8f acc[2];
    acc[0] = (v8f){0.f,0.f,0.f,0.f,0.f,0.f,0.f,0.f};
    acc[1] = acc[0];

    float4 wq[8];
    auto asyncA = [&](int kb, int b) {   // 16B of bf16 H -> LDS, async DMA
        unsigned long long ga = (unsigned long long)(uintptr_t)(hp + kb);
        asm volatile("global_load_async_to_lds_b128 %0, %1, off"
                     :: "v"(ldsX[b]), "v"(ga) : "memory");
    };
    auto loadW = [&](int kb) {
#pragma unroll
        for (int j = 0; j < 8; ++j) wq[j] = ((const float4*)(wp + kb))[j];
    };
    auto storeW = [&](int b) {
#pragma unroll
        for (int c = 0; c < 4; ++c) {
            v8bf vw;
#pragma unroll
            for (int j = 0; j < 2; ++j) {
                float4 q = wq[c*2+j];
                vw[j*4+0] = (__bf16)q.x; vw[j*4+1] = (__bf16)q.y;
                vw[j*4+2] = (__bf16)q.z; vw[j*4+3] = (__bf16)q.w;
            }
            *(v8bf*)&s_w[b][wr * LDSTR + wc + c * 8] = vw;
        }
    };

    asyncA(0, 0);
    loadW(0);
    storeW(0);
    asm volatile("s_wait_asynccnt 0" ::: "memory");
    __syncthreads();

    const int NS = D_FF / KC;
    for (int s = 0; s < NS; ++s) {
        const int b = s & 1;
        if (s + 1 < NS) {
            asyncA((s + 1) * KC, (s + 1) & 1);
            loadW((s + 1) * KC);
        }
#pragma unroll
        for (int ks = 0; ks < 2; ++ks) {
            const int kc = ks * 32;
            v8bf bb0 = *(const v8bf*)&s_w[b][(wv*16 + m) * LDSTR + kc + hi*8];
            v8bf bb1 = *(const v8bf*)&s_w[b][(wv*16 + m) * LDSTR + kc + 16 + hi*8];
            v16bf B = __builtin_shufflevector(bb0, bb1, 0,1,2,3,4,5,6,7,8,9,10,11,12,13,14,15);
#pragma unroll
            for (int mt = 0; mt < 2; ++mt) {
                v8bf aa0 = *(const v8bf*)&s_x[b][(mt*16 + m) * LDSTR + kc + hi*8];
                v8bf aa1 = *(const v8bf*)&s_x[b][(mt*16 + m) * LDSTR + kc + 16 + hi*8];
                v16bf A = __builtin_shufflevector(aa0, aa1, 0,1,2,3,4,5,6,7,8,9,10,11,12,13,14,15);
                acc[mt] = __builtin_amdgcn_wmma_f32_16x16x32_bf16(false, A, false, B,
                                                                 (short)0, acc[mt], false, false);
            }
        }
        if (s + 1 < NS) storeW((s + 1) & 1);
        asm volatile("s_wait_asynccnt 0" ::: "memory");
        __syncthreads();
    }

#pragma unroll
    for (int mt = 0; mt < 2; ++mt)
#pragma unroll
        for (int r = 0; r < 8; ++r) {
            const int M = mt * 16 + r + hi * 8;
            float v = acc[mt][r] * s_pw[M];
            atomicAdd(&out[(size_t)s_tok[M] * D_MODEL + dBase + wv * 16 + m], v);
        }
}

// ---------------------------------------------------------------------------
extern "C" void kernel_launch(void* const* d_in, const int* in_sizes, int n_in,
                              void* d_out, int out_size, void* d_ws, size_t ws_size,
                              hipStream_t stream) {
    const float* x  = (const float*)d_in[0];   // [2,2048,1024]
    const float* w1 = (const float*)d_in[1];   // [8,4096,1024]
    const float* w2 = (const float*)d_in[2];   // [8,1024,4096]
    const float* rw = (const float*)d_in[3];   // [8,1024]
    float* out = (float*)d_out;

    char* ws = (char*)d_ws;
    int* meta = (int*)ws;
    size_t off = 256;
    int*   topk_id  = (int*)(ws + off);   off += (size_t)NTOK * 2 * sizeof(int);
    float* topk_w   = (float*)(ws + off); off += (size_t)NTOK * 2 * sizeof(float);
    int*   pair_tok = (int*)(ws + off);   off += (size_t)NPAIR_CAP * sizeof(int);
    float* pair_w   = (float*)(ws + off); off += (size_t)NPAIR_CAP * sizeof(float);
    off = (off + 255) & ~(size_t)255;
    __bf16* Hbuf = (__bf16*)(ws + off);   // [NPAIR_CAP, D_FF] bf16

    hipMemsetAsync(ws, 0, 256, stream);
    hipMemsetAsync(d_out, 0, (size_t)out_size * sizeof(float), stream);

    k_router <<<NTOK / 8, 256, 0, stream>>>(x, rw, topk_id, topk_w, meta);
    k_scan   <<<1, 32, 0, stream>>>(meta, pair_tok, pair_w);
    k_scatter<<<NTOK / 256, 256, 0, stream>>>(topk_id, topk_w, meta, pair_tok, pair_w);
    k_gemm1  <<<dim3(NTOK / BM, NE, D_FF / BN), 256, 0, stream>>>(x, w1, meta, pair_tok, Hbuf);
    k_gemm2  <<<dim3(NTOK / BM, NE, D_MODEL / BN), 256, 0, stream>>>(Hbuf, w2, meta, pair_tok, pair_w, out);
}
